// SE_GNN_13134009991284
// MI455X (gfx1250) — matrix operand order repaired and checked by reading, not compiled, not run on hardware
//
#include <hip/hip_runtime.h>
#include <math.h>

#define H      200
#define HP     224     // H padded to multiple of 32 (WMMA K)
#define NENT   40000
#define NEDGE  400000
#define NLAYER 2
#define NREL   800
#define BSZ    512
#define OUTCH  200
#define FLAT   39200   // 14*14*200, multiple of 32

typedef __bf16 bf16;
typedef __attribute__((ext_vector_type(16))) __bf16 v16bf;
typedef __attribute__((ext_vector_type(8)))  __bf16 v8bf;
typedef __attribute__((ext_vector_type(8)))  float  v8f;

union frag_u { v16bf v; v8bf h[2]; };

// ---------------------------------------------------------------- utilities
__device__ inline void atomic_max_f32(float* addr, float val) {
  if (val >= 0.f) atomicMax((int*)addr, __float_as_int(val));
  else            atomicMin((unsigned int*)addr, __float_as_uint(val));
}

__global__ void fill_kernel(float* __restrict__ p, float v, int n) {
  int i = blockIdx.x * blockDim.x + threadIdx.x;
  if (i < n) p[i] = v;
}

// f32 [M][K] -> bf16 [M][Kp], zero padded columns (branchless select)
__global__ void pad_a_bf16_kernel(const float* __restrict__ in, bf16* __restrict__ out,
                                  int K, int Kp, long total) {
  long i = (long)blockIdx.x * blockDim.x + threadIdx.x;
  if (i >= total) return;
  int  k = (int)(i % Kp);
  long m = i / Kp;
  out[i] = (k < K) ? (bf16)in[m * K + k] : (bf16)0.f;
}

// B [K][N] f32 row-major -> WMMA-fragment-swizzled bf16:
//   out[((kc*ntiles + tn)*32 + lane)*16 + e] = B[kc*32 + (lane>>4)*16 + e][tn*16 + (lane&15)]
__global__ void swizzle_b_kernel(const float* __restrict__ B, bf16* __restrict__ out,
                                 int K, int N, int ntiles, long total) {
  long i = (long)blockIdx.x * blockDim.x + threadIdx.x;
  if (i >= total) return;
  int  e    = (int)(i & 15);
  int  lane = (int)((i >> 4) & 31);
  long tile = i >> 9;
  int  tn   = (int)(tile % ntiles);
  int  kc   = (int)(tile / ntiles);
  int  n = tn * 16 + (lane & 15);
  int  k = kc * 32 + ((lane >> 4) << 4) + e;
  out[i] = (k < K && n < N) ? (bf16)B[(long)k * N + n] : (bf16)0.f;
}

// B^T source: memory is [N][ldk] f32 (ent), logical B[k][n] = mem[n*ldk + k]
__global__ void swizzle_bt_kernel(const float* __restrict__ Bm, bf16* __restrict__ out,
                                  int K, int N, int ldk, int ntiles, long total) {
  long i = (long)blockIdx.x * blockDim.x + threadIdx.x;
  if (i >= total) return;
  int  e    = (int)(i & 15);
  int  lane = (int)((i >> 4) & 31);
  long tile = i >> 9;
  int  tn   = (int)(tile % ntiles);
  int  kc   = (int)(tile / ntiles);
  int  n = tn * 16 + (lane & 15);
  int  k = kc * 32 + ((lane >> 4) << 4) + e;
  out[i] = (k < K && n < N) ? (bf16)Bm[(long)n * ldk + k] : (bf16)0.f;
}

// ------------------------------------------------- fused edge-softmax passes
__global__ void edge_logit_kernel(const int* __restrict__ src, const int* __restrict__ dst,
                                  const int* __restrict__ rel_id,
                                  const float* __restrict__ ent, const float* __restrict__ rel_l,
                                  float* __restrict__ lg0, float* __restrict__ lg1, float* __restrict__ lg2,
                                  float* __restrict__ m0, float* __restrict__ m1, float* __restrict__ m2) {
  int wid  = (blockIdx.x * blockDim.x + threadIdx.x) >> 5;
  int lane = threadIdx.x & 31;
  if (wid >= NEDGE) return;
  int d = dst[wid];
  const float* v  = ent   + (long)d           * H;
  const float* u  = ent   + (long)src[wid]    * H;
  const float* er = rel_l + (long)rel_id[wid] * H;
  float a0 = 0.f, a1 = 0.f, a2 = 0.f;
  for (int i = lane; i < H; i += 32) {
    float uv = u[i], ev = er[i], vv = v[i];
    a0 += ev * vv;
    a1 += uv * vv;
    a2 += uv * ev * vv;
  }
  for (int off = 16; off; off >>= 1) {
    a0 += __shfl_down(a0, off, 32);
    a1 += __shfl_down(a1, off, 32);
    a2 += __shfl_down(a2, off, 32);
  }
  if (lane == 0) {
    lg0[wid] = a0; lg1[wid] = a1; lg2[wid] = a2;
    atomic_max_f32(&m0[d], a0);
    atomic_max_f32(&m1[d], a1);
    atomic_max_f32(&m2[d], a2);
  }
}

__global__ void edge_exp_kernel(const int* __restrict__ dst,
                                const float* __restrict__ m0, const float* __restrict__ m1, const float* __restrict__ m2,
                                float* __restrict__ lg0, float* __restrict__ lg1, float* __restrict__ lg2,
                                float* __restrict__ s0, float* __restrict__ s1, float* __restrict__ s2) {
  int e = blockIdx.x * blockDim.x + threadIdx.x;
  if (e >= NEDGE) return;
  int d = dst[e];
  float e0 = __expf(lg0[e] - m0[d]); lg0[e] = e0; atomicAdd(&s0[d], e0);
  float e1 = __expf(lg1[e] - m1[d]); lg1[e] = e1; atomicAdd(&s1[d], e1);
  float e2 = __expf(lg2[e] - m2[d]); lg2[e] = e2; atomicAdd(&s2[d], e2);
}

__global__ void edge_scatter_kernel(const int* __restrict__ src, const int* __restrict__ dst,
                                    const int* __restrict__ rel_id,
                                    const float* __restrict__ ent, const float* __restrict__ rel_l,
                                    const float* __restrict__ ex0, const float* __restrict__ ex1, const float* __restrict__ ex2,
                                    const float* __restrict__ s0, const float* __restrict__ s1, const float* __restrict__ s2,
                                    float* __restrict__ agg0, float* __restrict__ agg1, float* __restrict__ agg2) {
  int wid  = (blockIdx.x * blockDim.x + threadIdx.x) >> 5;
  int lane = threadIdx.x & 31;
  if (wid >= NEDGE) return;
  int d = dst[wid];
  float w0 = ex0[wid] / s0[d];
  float w1 = ex1[wid] / s1[d];
  float w2 = ex2[wid] / s2[d];
  const float* u  = ent   + (long)src[wid]    * H;
  const float* er = rel_l + (long)rel_id[wid] * H;
  long base = (long)d * H;
  for (int i = lane; i < H; i += 32) {
    float uv = u[i], ev = er[i];
    atomicAdd(&agg0[base + i], ev * w0);
    atomicAdd(&agg1[base + i], uv * w1);
    atomicAdd(&agg2[base + i], uv * ev * w2);
  }
}

// ---------------------------------------------------------- WMMA fragments
// A: bf16 [M][ldk] row-major, ldk multiple of 32, rows always in range.
// Per lane: two contiguous 16B chunks -> global_load_b128 x2, no predication.
__device__ inline v16bf load_a_frag(const bf16* __restrict__ A, int ldk,
                                    int row0, int k0, int lane) {
  const bf16* p = A + (long)(row0 + (lane & 15)) * ldk + k0 + ((lane >> 4) << 4);
  frag_u f;
  f.h[0] = *(const v8bf*)(p);
  f.h[1] = *(const v8bf*)(p + 8);
  return f.v;
}

// B: pre-swizzled fragments, 32 contiguous bytes per lane.
__device__ inline v16bf load_b_frag(const bf16* __restrict__ Bs, int ntiles,
                                    int kc, int tn, int lane) {
  const bf16* p = Bs + ((((long)kc * ntiles + tn) * 32 + lane) << 4);
  frag_u f;
  f.h[0] = *(const v8bf*)(p);
  f.h[1] = *(const v8bf*)(p + 8);
  return f.v;
}

// EPI 0: C += tanh(A@B)            (GNN layer update into ent)
// EPI 1: C  = sigmoid(A@B+bias[n]) (final scores)
template <int EPI>
__global__ void wmma_gemm_kernel(const bf16* __restrict__ A, const bf16* __restrict__ Bs,
                                 float* __restrict__ C, const float* __restrict__ bias,
                                 int M, int N, int Kp) {
  int wid  = (blockIdx.x * blockDim.x + threadIdx.x) >> 5;
  int lane = threadIdx.x & 31;
  int ntiles = (N + 15) >> 4;
  int mtiles = M >> 4;                       // M is a multiple of 16
  if (wid >= ntiles * mtiles) return;
  int row0 = (wid / ntiles) << 4;
  int tn   = wid % ntiles;
  v8f c = {0.f, 0.f, 0.f, 0.f, 0.f, 0.f, 0.f, 0.f};
  int nkc = Kp >> 5;
  for (int kc = 0; kc < nkc; ++kc) {
    v16bf a = load_a_frag(A, Kp, row0, kc << 5, lane);
    v16bf b = load_b_frag(Bs, ntiles, kc, tn, lane);
    c = __builtin_amdgcn_wmma_f32_16x16x32_bf16(false, a, false, b, (short)0, c, false, false);
  }
  int n = (tn << 4) + (lane & 15);
  if (n >= N) return;
#pragma unroll
  for (int r = 0; r < 8; ++r) {
    int m = row0 + r + ((lane >> 4) << 3);
    float v = c[r];
    long idx = (long)m * N + n;
    if (EPI == 0) C[idx] += tanhf(v);
    else          C[idx] = 1.f / (1.f + __expf(-(v + bias[n])));
  }
}

// Split-K (FC: K=39200, only 416 output tiles). kchunk counted in 32-wide k-steps.
__global__ void wmma_gemm_splitk_kernel(const bf16* __restrict__ A, const bf16* __restrict__ Bs,
                                        float* __restrict__ C, int M, int N, int Kp,
                                        int nk, int kchunk) {
  int wid  = (blockIdx.x * blockDim.x + threadIdx.x) >> 5;
  int lane = threadIdx.x & 31;
  int ntiles = (N + 15) >> 4;
  int mtiles = M >> 4;
  int tiles  = ntiles * mtiles;
  if (wid >= tiles * nk) return;
  int kcs  = wid / tiles;
  int tile = wid % tiles;
  int row0 = (tile / ntiles) << 4;
  int tn   = tile % ntiles;
  int nkc  = Kp >> 5;
  int kend = min(nkc, (kcs + 1) * kchunk);
  v8f c = {0.f, 0.f, 0.f, 0.f, 0.f, 0.f, 0.f, 0.f};
  for (int kc = kcs * kchunk; kc < kend; ++kc) {
    __builtin_prefetch(A + (long)(row0 + (lane & 15)) * Kp + (kc << 5) + 128, 0, 1);
    v16bf a = load_a_frag(A, Kp, row0, kc << 5, lane);
    v16bf b = load_b_frag(Bs, ntiles, kc, tn, lane);
    c = __builtin_amdgcn_wmma_f32_16x16x32_bf16(false, a, false, b, (short)0, c, false, false);
  }
  int n = (tn << 4) + (lane & 15);
  if (n >= N) return;
#pragma unroll
  for (int r = 0; r < 8; ++r) {
    int m = row0 + r + ((lane >> 4) << 3);
    atomicAdd(&C[(long)m * N + n], c[r]);
  }
}

// relu(part + fc_b) -> bf16 [BSZ][HP] with zero padding (A operand of score GEMM)
__global__ void fc_epilogue_kernel(const float* __restrict__ part, const float* __restrict__ fc_b,
                                   bf16* __restrict__ hbf, int total) {
  int i = blockIdx.x * blockDim.x + threadIdx.x;
  if (i >= total) return;
  int k = i % HP, m = i / HP;
  hbf[i] = (k < H) ? (bf16)fmaxf(part[m * H + k] + fc_b[k], 0.f) : (bf16)0.f;
}

// --------------------------------------------------------------- ConvE conv
__global__ void conve_conv_kernel(const int* __restrict__ h_id, const int* __restrict__ r_id,
                                  const float* __restrict__ ent, const float* __restrict__ pred_rel,
                                  const float* __restrict__ conv_w, bf16* __restrict__ out) {
  __shared__ float img[400];
  __shared__ float wsm[49];
  int b  = blockIdx.x / OUTCH;
  int oc = blockIdx.x % OUTCH;
  int t  = threadIdx.x;
  const float* head = ent      + (long)h_id[b] * H;
  const float* rel  = pred_rel + (long)r_id[b] * H;
  for (int i = t; i < 400; i += blockDim.x) img[i] = (i < 200) ? head[i] : rel[i - 200];
  for (int i = t; i < 49;  i += blockDim.x) wsm[i] = conv_w[oc * 49 + i];
  __syncthreads();
  for (int p = t; p < 196; p += blockDim.x) {
    int y = p / 14, x = p % 14;
    float acc = 0.f;
#pragma unroll
    for (int ky = 0; ky < 7; ++ky)
#pragma unroll
      for (int kx = 0; kx < 7; ++kx)
        acc += img[(y + ky) * 20 + (x + kx)] * wsm[ky * 7 + kx];
    out[(((long)b * OUTCH + oc) * 14 + y) * 14 + x] = (bf16)fmaxf(acc, 0.f);
  }
}

// ------------------------------------------------------------------- driver
static inline int cdiv(long a, long b) { return (int)((a + b - 1) / b); }

extern "C" void kernel_launch(void* const* d_in, const int* in_sizes, int n_in,
                              void* d_out, int out_size, void* d_ws, size_t ws_size,
                              hipStream_t stream) {
  (void)in_sizes; (void)n_in; (void)out_size; (void)ws_size;
  const int*   h_id     = (const int*)d_in[0];
  const int*   r_id     = (const int*)d_in[1];
  const int*   src      = (const int*)d_in[2];
  const int*   dst      = (const int*)d_in[3];
  const int*   rel_id   = (const int*)d_in[4];
  const float* ent_in   = (const float*)d_in[5];
  const float* rel_embs = (const float*)d_in[6];
  const float* edge_w   = (const float*)d_in[7];
  const float* node_w   = (const float*)d_in[8];
  const float* comp_w   = (const float*)d_in[9];
  const float* pred_rel = (const float*)d_in[10];
  const float* conv_w   = (const float*)d_in[11];
  const float* fc_w     = (const float*)d_in[12];
  const float* fc_b     = (const float*)d_in[13];
  const float* bias     = (const float*)d_in[14];
  float* out = (float*)d_out;

  char* ws = (char*)d_ws;                         // all offsets 256B aligned
  float* ent    = (float*)(ws + 0);               // [40000,200] f32      32.0 MB
  float* agg    = (float*)(ws + 32000000);        // [3][40000,200] f32   96.0 MB
  float* lg     = (float*)(ws + 128000000);       // [3][E] f32            4.8 MB
  float* mmx    = (float*)(ws + 132800000);       // [3][N]
  float* ssm    = (float*)(ws + 133280000);       // [3][N]
  bf16*  abf    = (bf16*) (ws + 133760000);       // [40000,224] bf16     17.9 MB
  bf16*  wswz   = (bf16*) (ws + 151680000);       // 7*13*512 bf16 swizzled W
  bf16*  entswz = (bf16*) (ws + 151773184);       // 7*2500*512 bf16      17.9 MB
  bf16*  fcwswz = (bf16*) (ws + 169693184);       // 1225*13*512 bf16     16.3 MB
  bf16*  convbf = (bf16*) (ws + 186000384);       // [512,39200] bf16     40.1 MB
  float* fcpart = (float*)(ws + 226141184);       // [512,200] f32
  bf16*  hbf    = (bf16*) (ws + 226550784);       // [512,224] bf16

  hipMemcpyAsync(ent, ent_in, (size_t)NENT * H * 4, hipMemcpyDeviceToDevice, stream);
  // swizzle fc weights [39200,200] -> fragment layout (1225 k-chunks x 13 n-tiles)
  long fcw_total = 1225L * 13 * 512;
  swizzle_b_kernel<<<cdiv(fcw_total, 256), 256, 0, stream>>>(fc_w, fcwswz, FLAT, H, 13, fcw_total);

  const float* Wsrc[3] = {edge_w, node_w, comp_w};
  int edgeWaveBlocks = cdiv((long)NEDGE * 32, 256);

  for (int l = 0; l < NLAYER; ++l) {
    const float* rel_l = rel_embs + (long)l * NREL * H;
    fill_kernel<<<cdiv(3L * NENT, 256), 256, 0, stream>>>(mmx, -3.4e38f, 3 * NENT);
    fill_kernel<<<cdiv(3L * NENT, 256), 256, 0, stream>>>(ssm, 0.f, 3 * NENT);
    fill_kernel<<<cdiv(3L * NENT * H, 256), 256, 0, stream>>>(agg, 0.f, 3 * NENT * H);

    edge_logit_kernel<<<edgeWaveBlocks, 256, 0, stream>>>(
        src, dst, rel_id, ent, rel_l,
        lg, lg + NEDGE, lg + 2 * NEDGE, mmx, mmx + NENT, mmx + 2 * NENT);
    edge_exp_kernel<<<cdiv(NEDGE, 256), 256, 0, stream>>>(
        dst, mmx, mmx + NENT, mmx + 2 * NENT,
        lg, lg + NEDGE, lg + 2 * NEDGE, ssm, ssm + NENT, ssm + 2 * NENT);
    edge_scatter_kernel<<<edgeWaveBlocks, 256, 0, stream>>>(
        src, dst, rel_id, ent, rel_l,
        lg, lg + NEDGE, lg + 2 * NEDGE, ssm, ssm + NENT, ssm + 2 * NENT,
        agg, agg + (long)NENT * H, agg + 2L * NENT * H);

    for (int mode = 0; mode < 3; ++mode) {
      long a_total = (long)NENT * HP;
      pad_a_bf16_kernel<<<cdiv(a_total, 256), 256, 0, stream>>>(
          agg + (long)mode * NENT * H, abf, H, HP, a_total);
      long w_total = 7L * 13 * 512;
      swizzle_b_kernel<<<cdiv(w_total, 256), 256, 0, stream>>>(
          Wsrc[mode] + (long)l * H * H, wswz, H, H, 13, w_total);
      int waves = (NENT / 16) * 13;               // 2500 * 13
      wmma_gemm_kernel<0><<<cdiv((long)waves * 32, 256), 256, 0, stream>>>(
          abf, wswz, ent, nullptr, NENT, H, HP);  // ent += tanh(agg @ W)
    }
  }

  // ConvE scoring
  long es_total = 7L * 2500 * 512;                // ent^T swizzled (B of score GEMM)
  swizzle_bt_kernel<<<cdiv(es_total, 256), 256, 0, stream>>>(
      ent, entswz, H, NENT, H, 2500, es_total);
  conve_conv_kernel<<<BSZ * OUTCH, 256, 0, stream>>>(h_id, r_id, ent, pred_rel, conv_w, convbf);

  fill_kernel<<<cdiv(BSZ * H, 256), 256, 0, stream>>>(fcpart, 0.f, BSZ * H);
  int nk = 25, kchunk = 49;                       // 25 * 49 = 1225 k-steps of 32
  int tiles = (BSZ / 16) * 13;                    // 32 * 13
  wmma_gemm_splitk_kernel<<<cdiv((long)tiles * nk * 32, 256), 256, 0, stream>>>(
      convbf, fcwswz, fcpart, BSZ, H, FLAT, nk, kchunk);
  fc_epilogue_kernel<<<cdiv(BSZ * HP, 256), 256, 0, stream>>>(fcpart, fc_b, hbf, BSZ * HP);

  int swaves = (BSZ / 16) * 2500;                 // 32 * 2500
  wmma_gemm_kernel<1><<<cdiv((long)swaves * 32, 256), 256, 0, stream>>>(
      hbf, entswz, out, bias, BSZ, NENT, HP);     // sigmoid(h @ ent^T + bias)
}
